// GCN_79731772883018
// MI455X (gfx1250) — matrix operand order repaired and checked by reading, not compiled, not run on hardware
//
#include <hip/hip_runtime.h>
#include <math.h>

typedef _Float16 h8   __attribute__((ext_vector_type(8)));
typedef _Float16 v16h __attribute__((ext_vector_type(16)));
typedef float    v8f  __attribute__((ext_vector_type(8)));

#define SHUF16(x,y) __builtin_shufflevector((x),(y),0,1,2,3,4,5,6,7,8,9,10,11,12,13,14,15)
#define WMMA(a,b,c) __builtin_amdgcn_wmma_f32_16x16x32_f16(false,(a),false,(b),(short)0,(c),false,false)

// ---------------- utility kernels ----------------

__global__ void zero_f32(float* __restrict__ p, size_t n) {
    size_t i = (size_t)blockIdx.x * blockDim.x + threadIdx.x;
    if (i < n) p[i] = 0.f;
}

__global__ void f32_to_f16(const float* __restrict__ a, _Float16* __restrict__ b, size_t n) {
    size_t i = (size_t)blockIdx.x * blockDim.x + threadIdx.x;
    if (i < n) b[i] = (_Float16)a[i];
}

// W [K,Nd] fp32 row-major -> Wt [Nd,K] f16 row-major (transposed)
__global__ void conv_transpose(const float* __restrict__ W, _Float16* __restrict__ Wt,
                               int K, int Nd) {
    int i = blockIdx.x * blockDim.x + threadIdx.x;
    if (i >= K * Nd) return;
    int k = i / Nd, n = i % Nd;
    Wt[(size_t)n * K + k] = (_Float16)W[i];
}

__global__ void deg_count(const int* __restrict__ dst, float* __restrict__ deg, int E) {
    int i = blockIdx.x * blockDim.x + threadIdx.x;
    if (i < E) atomicAdd(&deg[dst[i]], 1.0f);
}

__global__ void deg_finalize(float* __restrict__ deg, int Nn) {
    int i = blockIdx.x * blockDim.x + threadIdx.x;
    if (i < Nn) deg[i] = rsqrtf(deg[i] + 1.0f);   // +1 for self loop; in-place -> dinv
}

__global__ void norm_kernel(const int* __restrict__ src, const int* __restrict__ dst,
                            const float* __restrict__ dinv, float* __restrict__ norm, int E) {
    int i = blockIdx.x * blockDim.x + threadIdx.x;
    if (i < E) norm[i] = dinv[src[i]] * dinv[dst[i]];
}

// ---------------- WMMA GEMM: C[M,Nd] = A[M,K](f16) @ Bt[Nd,K]^T, fp32 out ----------
// one wave computes a 16x64 strip (4 N-tiles)
__global__ __launch_bounds__(256)
void gemm_wmma_f16(const _Float16* __restrict__ A, const _Float16* __restrict__ Bt,
                   float* __restrict__ C, int M, int K, int Nd, int totalTiles) {
    int lane = threadIdx.x & 31;
    int wv = threadIdx.x >> 5;
    int tile = blockIdx.x * (blockDim.x >> 5) + wv;
    if (tile >= totalTiles) return;
    int stripsN = Nd >> 6;
    int tM = tile / stripsN, tS = tile % stripsN;
    int ln = lane & 15, hi = lane >> 4;
    int ka = hi << 3;    // A-fragment K sub-offset (0 or 8)
    int kb = hi << 3;    // B-fragment K sub-offset (0 or 8)
    const _Float16* arow = A + (size_t)(tM * 16 + ln) * K;
    v8f z = {0.f,0.f,0.f,0.f,0.f,0.f,0.f,0.f};
    v8f acc[4];
#pragma unroll
    for (int i = 0; i < 4; i++) acc[i] = z;
    for (int k0 = 0; k0 < K; k0 += 32) {
        h8 alo = *(const h8*)(arow + k0 + ka);
        h8 ahi = *(const h8*)(arow + k0 + 16 + ka);
        v16h a = SHUF16(alo, ahi);
#pragma unroll
        for (int nt = 0; nt < 4; nt++) {
            const _Float16* brow = Bt + (size_t)(tS * 64 + nt * 16 + ln) * K;
            v16h b = SHUF16(*(const h8*)(brow + k0 + kb), *(const h8*)(brow + k0 + 16 + kb));
            acc[nt] = WMMA(a, b, acc[nt]);
        }
    }
#pragma unroll
    for (int nt = 0; nt < 4; nt++) {
        int n = tS * 64 + nt * 16 + ln;
#pragma unroll
        for (int r = 0; r < 8; r++)
            C[(size_t)(tM * 16 + hi * 8 + r) * Nd + n] = acc[nt][r];
    }
}

// ---------------- GCN scatter + finalize ----------------

__global__ void gcn_scatter(const float* __restrict__ xw, const int* __restrict__ src,
                            const int* __restrict__ dst, const float* __restrict__ norm,
                            float* __restrict__ acc, size_t total, int kmask, int kshift) {
    size_t gid = (size_t)blockIdx.x * blockDim.x + threadIdx.x;
    if (gid >= total) return;
    int e = (int)(gid >> kshift);
    int k = (int)(gid & (size_t)kmask);
    float v = xw[((size_t)src[e] << kshift) + k] * norm[e];
    atomicAdd(&acc[((size_t)dst[e] << kshift) + k], v);
}

__global__ void gcn_finalize(const float* __restrict__ acc, const float* __restrict__ xw,
                             const float* __restrict__ dinv, const float* __restrict__ bias,
                             _Float16* __restrict__ hout, size_t total, int kmask, int kshift) {
    size_t gid = (size_t)blockIdx.x * blockDim.x + threadIdx.x;
    if (gid >= total) return;
    int n = (int)(gid >> kshift);
    int k = (int)(gid & (size_t)kmask);
    float di = dinv[n];
    float v = acc[gid] + xw[gid] * di * di + bias[k];
    v = v > 0.f ? v : 0.01f * v;
    hout[gid] = (_Float16)v;
}

// ------- fused edge-MLP tile GEMM helper: 2 M-tiles (32 rows) from LDS A --------
// B fragments are loaded once and reused by both M-tiles.
template<int K, int ND, bool LRELU>
__device__ __forceinline__ void tile_gemm_lds2(const _Float16* Als,
                                               const _Float16* __restrict__ Bt,
                                               const float* __restrict__ bias,
                                               _Float16* Ols, int lane) {
    const int ln = lane & 15, hi = lane >> 4;
    const int ka = hi << 3, kb = hi << 3;
    v8f z = {0.f,0.f,0.f,0.f,0.f,0.f,0.f,0.f};
    v8f acc[2][ND / 16];
#pragma unroll
    for (int m = 0; m < 2; m++)
#pragma unroll
        for (int i = 0; i < ND / 16; i++) acc[m][i] = z;
#pragma unroll
    for (int k0 = 0; k0 < K; k0 += 32) {
        h8 a0lo = *(const h8*)(Als + ln * K + k0 + ka);
        h8 a0hi = *(const h8*)(Als + ln * K + k0 + 16 + ka);
        h8 a1lo = *(const h8*)(Als + (16 + ln) * K + k0 + ka);
        h8 a1hi = *(const h8*)(Als + (16 + ln) * K + k0 + 16 + ka);
        v16h a0 = SHUF16(a0lo, a0hi);
        v16h a1 = SHUF16(a1lo, a1hi);
#pragma unroll
        for (int nt = 0; nt < ND / 16; nt++) {
            v16h b = SHUF16(*(const h8*)(Bt + (size_t)(nt * 16 + ln) * K + k0 + kb), *(const h8*)(Bt + (size_t)(nt * 16 + ln) * K + k0 + 16 + kb));
            acc[0][nt] = WMMA(a0, b, acc[0][nt]);
            acc[1][nt] = WMMA(a1, b, acc[1][nt]);
        }
    }
#pragma unroll
    for (int nt = 0; nt < ND / 16; nt++) {
        int n = nt * 16 + ln;
        float bv = bias[n];
#pragma unroll
        for (int r = 0; r < 8; r++) {
            float v0 = acc[0][nt][r] + bv;
            float v1 = acc[1][nt][r] + bv;
            if (LRELU) {
                v0 = v0 > 0.f ? v0 : 0.01f * v0;
                v1 = v1 > 0.f ? v1 : 0.01f * v1;
            }
            Ols[(hi * 8 + r) * ND + n] = (_Float16)v0;
            Ols[(16 + hi * 8 + r) * ND + n] = (_Float16)v1;
        }
    }
}

// ------- fused edge kernel: encoder + 576->128->64 + 2 heads; 32 edges/wave ------
__global__ __launch_bounds__(64)
void GCN_edge_head(const _Float16* __restrict__ h3, const int* __restrict__ src,
                   const int* __restrict__ dst, const float* __restrict__ eattr,
                   const float* __restrict__ We, const float* __restrict__ be,
                   const _Float16* __restrict__ Wf1t, const float* __restrict__ bf1,
                   const _Float16* __restrict__ Wf2t, const float* __restrict__ bf2,
                   const _Float16* __restrict__ Wc1t, const float* __restrict__ bc1,
                   const float* __restrict__ Wc2, const float* __restrict__ bc2,
                   const _Float16* __restrict__ Wr1t, const float* __restrict__ br1,
                   const float* __restrict__ Wr2, const float* __restrict__ br2,
                   float* __restrict__ out, int E) {
    __shared__ float We_s[16 * 64];
    __shared__ float be_s[64];
    __shared__ float attr_s[2][32 * 16];
    __shared__ _Float16 enc_s[2][32 * 64];
    __shared__ _Float16 hid_s[2][32 * 128];
    __shared__ _Float16 com_s[2][32 * 64];
    __shared__ float tr_s[2][32];

    int tid = threadIdx.x;
    for (int i = tid; i < 16 * 64; i += 64) We_s[i] = We[i];
    if (tid < 64) be_s[tid] = be[tid];
    __syncthreads();

    int lane = tid & 31, wv = tid >> 5;
    int tile = blockIdx.x * 2 + wv;
    int e0 = tile * 32;
    if (e0 >= E) return;                 // wave-uniform exit
    int ln = lane & 15, hi = lane >> 4;
    int ka = hi << 3, kb = hi << 3;
    int ea = e0 + ln, eb = e0 + 16 + ln;
    const _Float16* hs0 = h3 + (size_t)src[ea] * 256;
    const _Float16* hd0 = h3 + (size_t)dst[ea] * 256;
    const _Float16* hs1 = h3 + (size_t)src[eb] * 256;
    const _Float16* hd1 = h3 + (size_t)dst[eb] * 256;

    // stage edge_attr (32 edges), compute edge encoder (32x16 @ 16x64) into LDS f16
    for (int i = lane; i < 512; i += 32) attr_s[wv][i] = eattr[(size_t)e0 * 16 + i];
    for (int idx = lane; idx < 2048; idx += 32) {
        int mm = idx >> 6, n = idx & 63;
        float a = be_s[n];
#pragma unroll
        for (int k = 0; k < 16; k++) a += attr_s[wv][mm * 16 + k] * We_s[k * 64 + n];
        a = a > 0.f ? a : 0.01f * a;
        enc_s[wv][mm * 64 + n] = (_Float16)a;
    }

    // f1: combined[32,576] @ Wf1[576,128]; A gathered on the fly; B reused by 2 M-tiles
    v8f z = {0.f,0.f,0.f,0.f,0.f,0.f,0.f,0.f};
    v8f acc1[2][8];
#pragma unroll
    for (int m = 0; m < 2; m++)
#pragma unroll
        for (int i = 0; i < 8; i++) acc1[m][i] = z;

#pragma unroll
    for (int k0 = 0; k0 < 256; k0 += 32) {                 // k in [0,256): h3[src]
        h8 a0lo = *(const h8*)(hs0 + k0 + ka);
        h8 a0hi = *(const h8*)(hs0 + k0 + 16 + ka);
        h8 a1lo = *(const h8*)(hs1 + k0 + ka);
        h8 a1hi = *(const h8*)(hs1 + k0 + 16 + ka);
        v16h a0 = SHUF16(a0lo, a0hi);
        v16h a1 = SHUF16(a1lo, a1hi);
#pragma unroll
        for (int nt = 0; nt < 8; nt++) {
            v16h b = SHUF16(*(const h8*)(Wf1t + (size_t)(nt * 16 + ln) * 576 + k0 + kb), *(const h8*)(Wf1t + (size_t)(nt * 16 + ln) * 576 + k0 + 16 + kb));
            acc1[0][nt] = WMMA(a0, b, acc1[0][nt]);
            acc1[1][nt] = WMMA(a1, b, acc1[1][nt]);
        }
    }
#pragma unroll
    for (int k0 = 0; k0 < 256; k0 += 32) {                 // k in [256,512): h3[dst]
        h8 a0lo = *(const h8*)(hd0 + k0 + ka);
        h8 a0hi = *(const h8*)(hd0 + k0 + 16 + ka);
        h8 a1lo = *(const h8*)(hd1 + k0 + ka);
        h8 a1hi = *(const h8*)(hd1 + k0 + 16 + ka);
        v16h a0 = SHUF16(a0lo, a0hi);
        v16h a1 = SHUF16(a1lo, a1hi);
#pragma unroll
        for (int nt = 0; nt < 8; nt++) {
            v16h b = SHUF16(*(const h8*)(Wf1t + (size_t)(nt * 16 + ln) * 576 + 256 + k0 + kb), *(const h8*)(Wf1t + (size_t)(nt * 16 + ln) * 576 + 256 + k0 + 16 + kb));
            acc1[0][nt] = WMMA(a0, b, acc1[0][nt]);
            acc1[1][nt] = WMMA(a1, b, acc1[1][nt]);
        }
    }
#pragma unroll
    for (int k0 = 0; k0 < 64; k0 += 32) {                  // k in [512,576): edge enc
        h8 a0lo = *(const h8*)(&enc_s[wv][ln * 64 + k0 + ka]);
        h8 a0hi = *(const h8*)(&enc_s[wv][ln * 64 + k0 + 16 + ka]);
        h8 a1lo = *(const h8*)(&enc_s[wv][(16 + ln) * 64 + k0 + ka]);
        h8 a1hi = *(const h8*)(&enc_s[wv][(16 + ln) * 64 + k0 + 16 + ka]);
        v16h a0 = SHUF16(a0lo, a0hi);
        v16h a1 = SHUF16(a1lo, a1hi);
#pragma unroll
        for (int nt = 0; nt < 8; nt++) {
            v16h b = SHUF16(*(const h8*)(Wf1t + (size_t)(nt * 16 + ln) * 576 + 512 + k0 + kb), *(const h8*)(Wf1t + (size_t)(nt * 16 + ln) * 576 + 512 + k0 + 16 + kb));
            acc1[0][nt] = WMMA(a0, b, acc1[0][nt]);
            acc1[1][nt] = WMMA(a1, b, acc1[1][nt]);
        }
    }
#pragma unroll
    for (int nt = 0; nt < 8; nt++) {
        int n = nt * 16 + ln;
        float bv = bf1[n];
#pragma unroll
        for (int r = 0; r < 8; r++) {
            float v0 = acc1[0][nt][r] + bv;
            float v1 = acc1[1][nt][r] + bv;
            v0 = v0 > 0.f ? v0 : 0.01f * v0;
            v1 = v1 > 0.f ? v1 : 0.01f * v1;
            hid_s[wv][(hi * 8 + r) * 128 + n] = (_Float16)v0;
            hid_s[wv][(16 + hi * 8 + r) * 128 + n] = (_Float16)v1;
        }
    }

    // f2: [32,128]@[128,64] + bf2 (no activation) -> common
    tile_gemm_lds2<128, 64, false>(&hid_s[wv][0], Wf2t, bf2, &com_s[wv][0], lane);

    // classifier: lrelu(common@Wc1+bc1), then sigmoid(.@Wc2+bc2); one edge per lane
    tile_gemm_lds2<64, 64, true>(&com_s[wv][0], Wc1t, bc1, &hid_s[wv][0], lane);
    {
        float a = bc2[0];
#pragma unroll
        for (int k = 0; k < 64; k++) a += (float)hid_s[wv][lane * 64 + k] * Wc2[k];
        tr_s[wv][lane] = 1.f / (1.f + __expf(-a));
    }
    // regressor: lrelu(common@Wr1+br1), then .@Wr2+br2
    tile_gemm_lds2<64, 64, true>(&com_s[wv][0], Wr1t, br1, &hid_s[wv][0], lane);
    {
        float a = br2[0];
#pragma unroll
        for (int k = 0; k < 64; k++) a += (float)hid_s[wv][lane * 64 + k] * Wr2[k];
        float t = tr_s[wv][lane];
        out[e0 + lane] = t * a;        // final_pred
        out[E + e0 + lane] = t;        // trade_exists
    }
}

// ---------------- host orchestration ----------------
extern "C" void kernel_launch(void* const* d_in, const int* in_sizes, int n_in,
                              void* d_out, int out_size, void* d_ws, size_t ws_size,
                              hipStream_t stream) {
    const float* x     = (const float*)d_in[0];
    const int*   ei    = (const int*)d_in[1];
    const float* eattr = (const float*)d_in[2];
    const float* W1 = (const float*)d_in[3];   const float* b1 = (const float*)d_in[4];
    const float* W2 = (const float*)d_in[5];   const float* b2 = (const float*)d_in[6];
    const float* W3 = (const float*)d_in[7];   const float* b3 = (const float*)d_in[8];
    const float* We = (const float*)d_in[9];   const float* be = (const float*)d_in[10];
    const float* Wf1= (const float*)d_in[11];  const float* bf1= (const float*)d_in[12];
    const float* Wf2= (const float*)d_in[13];  const float* bf2= (const float*)d_in[14];
    const float* Wc1= (const float*)d_in[15];  const float* bc1= (const float*)d_in[16];
    const float* Wc2= (const float*)d_in[17];  const float* bc2= (const float*)d_in[18];
    const float* Wr1= (const float*)d_in[19];  const float* br1= (const float*)d_in[20];
    const float* Wr2= (const float*)d_in[21];  const float* br2= (const float*)d_in[22];
    (void)n_in; (void)out_size; (void)ws_size;

    const int IN = 128;
    const int Nn = in_sizes[0] / IN;     // 50000
    const int E  = in_sizes[1] / 2;      // 400000
    const int* srcp = ei;
    const int* dstp = ei + E;
    float* out = (float*)d_out;

    uintptr_t base = (uintptr_t)d_ws;
    size_t off = 0;
    auto alloc = [&](size_t bytes) -> void* {
        off = (off + 255) & ~(size_t)255;
        void* p = (void*)(base + off);
        off += bytes;
        return p;
    };
    float*     dinv = (float*)alloc((size_t)Nn * 4);
    float*     norm = (float*)alloc((size_t)E * 4);
    _Float16*  xh   = (_Float16*)alloc((size_t)Nn * 128 * 2);
    float*     xw   = (float*)alloc((size_t)Nn * 256 * 4);
    float*     accb = (float*)alloc((size_t)Nn * 256 * 4);
    _Float16*  h1   = (_Float16*)alloc((size_t)Nn * 64 * 2);
    _Float16*  h2   = (_Float16*)alloc((size_t)Nn * 128 * 2);
    _Float16*  h3   = (_Float16*)alloc((size_t)Nn * 256 * 2);
    _Float16*  W1t  = (_Float16*)alloc((size_t)64 * 128 * 2);
    _Float16*  W2t  = (_Float16*)alloc((size_t)128 * 64 * 2);
    _Float16*  W3t  = (_Float16*)alloc((size_t)256 * 128 * 2);
    _Float16*  Wf1t = (_Float16*)alloc((size_t)128 * 576 * 2);
    _Float16*  Wf2t = (_Float16*)alloc((size_t)64 * 128 * 2);
    _Float16*  Wc1t = (_Float16*)alloc((size_t)64 * 64 * 2);
    _Float16*  Wr1t = (_Float16*)alloc((size_t)64 * 64 * 2);

    const int TB = 256;
    auto cdiv = [](long long a, long long b) { return (int)((a + b - 1) / b); };

    // degrees / norms
    zero_f32<<<cdiv(Nn, TB), TB, 0, stream>>>(dinv, (size_t)Nn);
    deg_count<<<cdiv(E, TB), TB, 0, stream>>>(dstp, dinv, E);
    deg_finalize<<<cdiv(Nn, TB), TB, 0, stream>>>(dinv, Nn);
    norm_kernel<<<cdiv(E, TB), TB, 0, stream>>>(srcp, dstp, dinv, norm, E);

    // weight convert + transpose to f16
    conv_transpose<<<cdiv(128 * 64, TB), TB, 0, stream>>>(W1, W1t, 128, 64);
    conv_transpose<<<cdiv(64 * 128, TB), TB, 0, stream>>>(W2, W2t, 64, 128);
    conv_transpose<<<cdiv(128 * 256, TB), TB, 0, stream>>>(W3, W3t, 128, 256);
    conv_transpose<<<cdiv(576 * 128, TB), TB, 0, stream>>>(Wf1, Wf1t, 576, 128);
    conv_transpose<<<cdiv(128 * 64, TB), TB, 0, stream>>>(Wf2, Wf2t, 128, 64);
    conv_transpose<<<cdiv(64 * 64, TB), TB, 0, stream>>>(Wc1, Wc1t, 64, 64);
    conv_transpose<<<cdiv(64 * 64, TB), TB, 0, stream>>>(Wr1, Wr1t, 64, 64);

    f32_to_f16<<<cdiv((long long)Nn * 128, TB), TB, 0, stream>>>(x, xh, (size_t)Nn * 128);

    // GCN layer 1: 128 -> 64
    {
        int Kd = 64, sh = 6;
        int tiles = (Nn / 16) * (Kd / 64);
        gemm_wmma_f16<<<cdiv(tiles, 8), 256, 0, stream>>>(xh, W1t, xw, Nn, 128, Kd, tiles);
        zero_f32<<<cdiv((long long)Nn * Kd, TB), TB, 0, stream>>>(accb, (size_t)Nn * Kd);
        gcn_scatter<<<cdiv((long long)E * Kd, TB), TB, 0, stream>>>(xw, srcp, dstp, norm, accb,
                                                                    (size_t)E * Kd, Kd - 1, sh);
        gcn_finalize<<<cdiv((long long)Nn * Kd, TB), TB, 0, stream>>>(accb, xw, dinv, b1, h1,
                                                                      (size_t)Nn * Kd, Kd - 1, sh);
    }
    // GCN layer 2: 64 -> 128
    {
        int Kd = 128, sh = 7;
        int tiles = (Nn / 16) * (Kd / 64);
        gemm_wmma_f16<<<cdiv(tiles, 8), 256, 0, stream>>>(h1, W2t, xw, Nn, 64, Kd, tiles);
        zero_f32<<<cdiv((long long)Nn * Kd, TB), TB, 0, stream>>>(accb, (size_t)Nn * Kd);
        gcn_scatter<<<cdiv((long long)E * Kd, TB), TB, 0, stream>>>(xw, srcp, dstp, norm, accb,
                                                                    (size_t)E * Kd, Kd - 1, sh);
        gcn_finalize<<<cdiv((long long)Nn * Kd, TB), TB, 0, stream>>>(accb, xw, dinv, b2, h2,
                                                                      (size_t)Nn * Kd, Kd - 1, sh);
    }
    // GCN layer 3: 128 -> 256
    {
        int Kd = 256, sh = 8;
        int tiles = (Nn / 16) * (Kd / 64);
        gemm_wmma_f16<<<cdiv(tiles, 8), 256, 0, stream>>>(h2, W3t, xw, Nn, 128, Kd, tiles);
        zero_f32<<<cdiv((long long)Nn * Kd, TB), TB, 0, stream>>>(accb, (size_t)Nn * Kd);
        gcn_scatter<<<cdiv((long long)E * Kd, TB), TB, 0, stream>>>(xw, srcp, dstp, norm, accb,
                                                                    (size_t)E * Kd, Kd - 1, sh);
        gcn_finalize<<<cdiv((long long)Nn * Kd, TB), TB, 0, stream>>>(accb, xw, dinv, b3, h3,
                                                                      (size_t)Nn * Kd, Kd - 1, sh);
    }

    // fused per-edge MLP + heads (32 edges per wave, 2 waves per block)
    GCN_edge_head<<<cdiv((long long)E / 32, 2), 64, 0, stream>>>(
        h3, srcp, dstp, eattr, We, be, Wf1t, bf1, Wf2t, bf2,
        Wc1t, bc1, Wc2, bc2, Wr1t, br1, Wr2, br2, out, E);
}
